// PAAE_layer_23356032156040
// MI455X (gfx1250) — matrix operand order repaired and checked by reading, not compiled, not run on hardware
//
#include <hip/hip_runtime.h>
#include <hip/hip_bf16.h>
#include <cstdint>

typedef __attribute__((ext_vector_type(16))) _Float16 v16h;
typedef __attribute__((ext_vector_type(8)))  float    v8f;

#define U_DIM 2048
#define I_DIM 4096
#define D_DIM 64

// ---------------------------------------------------------------------------
// f32 -> f16 convert
// ---------------------------------------------------------------------------
__global__ __launch_bounds__(256) void cvt_f32_f16(const float* __restrict__ in,
                                                   _Float16* __restrict__ out, int n) {
    int i = blockIdx.x * 256 + threadIdx.x;
    if (i < n) out[i] = (_Float16)in[i];
}

// in [R,C] f32 -> out [C,R] f16 (32x32 tiles)
__global__ __launch_bounds__(256) void transpose_f32_f16(const float* __restrict__ in,
                                                         _Float16* __restrict__ out,
                                                         int R, int C) {
    __shared__ float t[32][33];
    const int bx = blockIdx.x * 32;   // col base in input
    const int by = blockIdx.y * 32;   // row base in input
    const int tx = threadIdx.x & 31;
    const int ty = threadIdx.x >> 5;  // 0..7
#pragma unroll
    for (int j = 0; j < 32; j += 8)
        t[ty + j][tx] = in[(size_t)(by + ty + j) * C + bx + tx];
    __syncthreads();
#pragma unroll
    for (int j = 0; j < 32; j += 8)
        out[(size_t)(bx + ty + j) * R + by + tx] = (_Float16)t[tx][ty + j];
}

// ---------------------------------------------------------------------------
// Tiled WMMA GEMM: C = act(A[MxK] * B + bias)
//   BLAYOUT==0: B stored [N,K] row-major (X @ W.T with W=[out,in])
//   BLAYOUT==1: B stored [K,N] row-major
//   ACT: 0=none 1=tanh(relu(x)) 2=tanh 3=sigmoid
// Double-buffered LDS, register-staged global->LDS pipeline, BK=32.
// Steady-state loop is branchless (last tile peeled).
// Requires M%BM==0, N%BN==0, K%32==0, K>=64 (true for all calls here).
// ---------------------------------------------------------------------------
template <int BM, int BN, int BLAYOUT, int ACT>
__global__ __launch_bounds__(256) void gemm_wmma(
    const _Float16* __restrict__ A, int lda,
    const _Float16* __restrict__ B, int ldb,
    const float* __restrict__ bias,
    _Float16* __restrict__ o16a, int ld16a,
    _Float16* __restrict__ o16b, int ld16b,
    float* __restrict__ o32, int ld32,
    int M, int N, int K) {
    constexpr int BK   = 32;
    constexpr int LDT  = BK + 8;                 // 40 halves = 80B (16B aligned)
    constexpr int WM   = BM / 4;                 // 4 waves along M
    constexpr int WN   = BN / 2;                 // 2 waves along N
    constexpr int MF   = WM / 16;
    constexpr int NF   = WN / 16;
    constexpr int A_IT = (BM * (BK / 8)) / 256;  // uint4 slots per thread for A
    constexpr int NQ   = BN / 8;
    constexpr int B_IT = (BLAYOUT == 0) ? (BN * (BK / 8)) / 256 : (BK * NQ) / 256;
    constexpr int KSTEP = 256 / NQ;              // KN-layout: k rows per staging iter

    __shared__ __align__(16) _Float16 lsA[2][BM * LDT];
    __shared__ __align__(16) _Float16 lsB[2][BN * LDT];

    const int tid    = threadIdx.x;
    const int lane   = tid & 31;
    const int wave   = tid >> 5;
    const int wm     = wave & 3;
    const int wn     = wave >> 2;
    const int laneR  = lane & 15;
    const int laneHi = lane >> 4;  // 0 or 1
    const int blockM = blockIdx.x * BM;
    const int blockN = blockIdx.y * BN;

    // staging coordinates (constexpr-trip, fully unrolled)
    const int aRow = tid >> 2;          // 0..63, +64 per iter
    const int aKq  = (tid & 3) << 3;    // 0..24 halves
    const int bKr  = tid / NQ;          // KN layout k-row
    const int bN0  = (tid % NQ) << 3;   // KN layout n offset

    const _Float16* aPtr  = A + (size_t)(blockM + aRow) * lda + aKq;
    const _Float16* bPtrT = B + (size_t)(blockN + aRow) * ldb + aKq;          // [N,K]
    const _Float16* bPtrN = B + (size_t)bKr * ldb + blockN + bN0;             // [K,N]

    uint4 ar[A_IT];
    uint4 br[B_IT];

    auto loadRegs = [&](int k0) {
#pragma unroll
        for (int i = 0; i < A_IT; ++i)
            ar[i] = *(const uint4*)(aPtr + (size_t)k0 + (size_t)i * 64 * lda);
        if (BLAYOUT == 0) {
#pragma unroll
            for (int i = 0; i < B_IT; ++i)
                br[i] = *(const uint4*)(bPtrT + (size_t)k0 + (size_t)i * 64 * ldb);
        } else {
#pragma unroll
            for (int i = 0; i < B_IT; ++i)
                br[i] = *(const uint4*)(bPtrN + (size_t)(k0 + i * KSTEP) * ldb);
        }
    };
    auto storeRegs = [&](int buf) {
#pragma unroll
        for (int i = 0; i < A_IT; ++i)
            *(uint4*)&lsA[buf][(aRow + i * 64) * LDT + aKq] = ar[i];
        if (BLAYOUT == 0) {
#pragma unroll
            for (int i = 0; i < B_IT; ++i)
                *(uint4*)&lsB[buf][(aRow + i * 64) * LDT + aKq] = br[i];
        } else {
#pragma unroll
            for (int i = 0; i < B_IT; ++i) {
                union { uint4 q; _Float16 h[8]; } u;
                u.q = br[i];
#pragma unroll
                for (int j = 0; j < 8; ++j)
                    lsB[buf][(bN0 + j) * LDT + bKr + i * KSTEP] = u.h[j];
            }
        }
    };

    v8f acc[MF][NF] = {};

    auto compute = [&](int buf) {
        // A 16x32 f16: lanes<16 -> M=lane, K={0..7,16..23}; lanes>=16 -> K={8..15,24..31}
        v16h af[MF];
#pragma unroll
        for (int i = 0; i < MF; ++i) {
            const int off = (wm * WM + i * 16 + laneR) * LDT + laneHi * 8;
            union { uint4 q[2]; v16h v; } u;
            u.q[0] = *(const uint4*)&lsA[buf][off];
            u.q[1] = *(const uint4*)&lsA[buf][off + 16];
            af[i] = u.v;
        }
        // B 32x16 f16: lanes<16 -> N=lane, K={0..7,16..23}; lanes>=16 -> K={8..15,24..31}
        v16h bf[NF];
#pragma unroll
        for (int j = 0; j < NF; ++j) {
            const int off = (wn * WN + j * 16 + laneR) * LDT + laneHi * 8;
            union { uint4 q[2]; v16h v; } u;
            u.q[0] = *(const uint4*)&lsB[buf][off];
            u.q[1] = *(const uint4*)&lsB[buf][off + 16];
            bf[j] = u.v;
        }
#pragma unroll
        for (int i = 0; i < MF; ++i)
#pragma unroll
            for (int j = 0; j < NF; ++j)
                acc[i][j] = __builtin_amdgcn_wmma_f32_16x16x32_f16(
                    false, af[i], false, bf[j], (short)0, acc[i][j], false, false);
    };

    // prologue: stage tile 0
    loadRegs(0);
    storeRegs(0);
    __syncthreads();

    // branchless steady-state: load tile k, compute tile k-1, store tile k
    int cur = 0;
    for (int k0 = BK; k0 < K; k0 += BK) {
        loadRegs(k0);                                   // all global loads, no wait
        __builtin_prefetch(aPtr + (size_t)(k0 + BK), 0, 3);  // speculative tile+1
        compute(cur);                                   // ds_load + wmma on buf cur
        storeRegs(cur ^ 1);                             // drain regs -> other buffer
        __syncthreads();
        cur ^= 1;
    }
    // peeled final tile
    compute(cur);

    // ---- epilogue: C/D layout VGPR r: M = r + 8*laneHi, N = laneR ----
#pragma unroll
    for (int i = 0; i < MF; ++i) {
#pragma unroll
        for (int j = 0; j < NF; ++j) {
            const int gCol = blockN + wn * WN + j * 16 + laneR;
            const float bv = bias ? bias[gCol] : 0.0f;
#pragma unroll
            for (int r = 0; r < 8; ++r) {
                const int gRow = blockM + wm * WM + i * 16 + r + laneHi * 8;
                float v = acc[i][j][r] + bv;
                if (ACT == 1)      v = (v > 0.0f) ? tanhf(v) : 0.0f;
                else if (ACT == 2) v = tanhf(v);
                else if (ACT == 3) v = 1.0f / (1.0f + __expf(-v));
                if (o16a) o16a[(size_t)gRow * ld16a + gCol] = (_Float16)v;
                if (o16b) o16b[(size_t)gRow * ld16b + gCol] = (_Float16)v;
                if (o32)  o32[(size_t)gRow * ld32 + gCol]   = v;
            }
        }
    }
}

// ---------------------------------------------------------------------------
// Row softmax: S[row, 0..cols) f32 -> O f16. One block (256 thr) per row.
// ---------------------------------------------------------------------------
__global__ __launch_bounds__(256) void softmax_rows(const float* __restrict__ S,
                                                    _Float16* __restrict__ O, int cols) {
    const int row = blockIdx.x;
    const int tid = threadIdx.x;
    const float* src = S + (size_t)row * cols;
    __shared__ float red[8];
    float v[16];
    int cnt = 0;
    float mx = -3.4e38f;
    for (int c = tid; c < cols; c += 256) {
        float x = src[c];
        v[cnt++] = x;
        mx = fmaxf(mx, x);
    }
#pragma unroll
    for (int o = 16; o > 0; o >>= 1) mx = fmaxf(mx, __shfl_xor(mx, o, 32));
    if ((tid & 31) == 0) red[tid >> 5] = mx;
    __syncthreads();
    float m = red[0];
#pragma unroll
    for (int i = 1; i < 8; ++i) m = fmaxf(m, red[i]);
    float sum = 0.0f;
    for (int i = 0; i < cnt; ++i) { v[i] = __expf(v[i] - m); sum += v[i]; }
#pragma unroll
    for (int o = 16; o > 0; o >>= 1) sum += __shfl_xor(sum, o, 32);
    __syncthreads();
    if ((tid & 31) == 0) red[tid >> 5] = sum;
    __syncthreads();
    float s = 0.0f;
#pragma unroll
    for (int i = 0; i < 8; ++i) s += red[i];
    const float inv = 1.0f / s;
    cnt = 0;
    for (int c = tid; c < cols; c += 256)
        O[(size_t)row * cols + c] = (_Float16)(v[cnt++] * inv);
}

// rate = sigmoid(u_mp + u_pmp); out16 = rate*u_mp + (1-rate)*u_pmp
__global__ __launch_bounds__(256) void fuse_users(const float* __restrict__ a,
                                                  const float* __restrict__ b,
                                                  _Float16* __restrict__ out, int n) {
    int i = blockIdx.x * 256 + threadIdx.x;
    if (i < n) {
        float um = a[i], up = b[i];
        float r = 1.0f / (1.0f + __expf(-(um + up)));
        out[i] = (_Float16)(r * um + (1.0f - r) * up);
    }
}

// ---------------------------------------------------------------------------
// Host orchestration
// ---------------------------------------------------------------------------
static inline size_t align_up(size_t x, size_t a) { return (x + a - 1) & ~(a - 1); }

extern "C" void kernel_launch(void* const* d_in, const int* in_sizes, int n_in,
                              void* d_out, int out_size, void* d_ws, size_t ws_size,
                              hipStream_t stream) {
    (void)in_sizes; (void)n_in; (void)out_size; (void)ws_size;
    const int U = U_DIM, I = I_DIM, D = D_DIM, UI = U_DIM + I_DIM;

    // --- bump allocator over workspace (scratch reused across branches) ---
    char* base = (char*)d_ws;
    size_t off = 0;
    auto alloc = [&](size_t bytes) -> char* {
        char* p = base + off;
        off = align_up(off + bytes, 256);
        return p;
    };
    _Float16* ui16   = (_Float16*)alloc((size_t)U * I * 2);
    _Float16* uiT16  = (_Float16*)alloc((size_t)I * U * 2);
    _Float16* Wu16   = (_Float16*)alloc((size_t)D * I * 2);
    _Float16* Wi16   = (_Float16*)alloc((size_t)D * U * 2);
    _Float16* mw16   = (_Float16*)alloc((size_t)I * D * 2);
    _Float16* aw16   = (_Float16*)alloc((size_t)I * I * 2);
    _Float16* Wat16  = (_Float16*)alloc((size_t)U * UI * 2);
    _Float16* x16    = (_Float16*)alloc((size_t)UI * D * 2);
    _Float16* h16    = (_Float16*)alloc((size_t)UI * I * 2);
    _Float16* a16    = (_Float16*)alloc((size_t)UI * I * 2);
    float*    scores = (float*)   alloc((size_t)U * I * 4);
    _Float16* at16   = (_Float16*)alloc((size_t)U * I * 2);
    _Float16* catU0  = (_Float16*)alloc((size_t)U * 2 * D * 2);
    _Float16* catU1  = (_Float16*)alloc((size_t)U * 2 * D * 2);
    _Float16* catIE  = (_Float16*)alloc((size_t)I * 2 * D * 2);
    float*    uf0    = (float*)   alloc((size_t)U * D * 4);
    float*    uf1    = (float*)   alloc((size_t)U * D * 4);
    _Float16* fuse16 = (_Float16*)alloc((size_t)U * D * 2);
    _Float16* mlpw16 = (_Float16*)alloc((size_t)D * 2 * D * 2);
    _Float16* trw16  = (_Float16*)alloc((size_t)D * D * 2);

    const float* mlp_b = (const float*)d_in[17];
    const float* tr_b  = (const float*)d_in[19];
    float* out = (float*)d_out;  // [U*D user_embeds][I*D item_embeddings]

    auto cvt = [&](const float* src, _Float16* dst, int n) {
        cvt_f32_f16<<<(n + 255) / 256, 256, 0, stream>>>(src, dst, n);
    };
    cvt((const float*)d_in[16], mlpw16, D * 2 * D);
    cvt((const float*)d_in[18], trw16, D * D);

    _Float16* catU[2] = { catU0, catU1 };
    float*    ufb[2]  = { uf0, uf1 };

    for (int b = 0; b < 2; ++b) {
        const float* ui  = (const float*)d_in[0 + b];
        const float* Wu  = (const float*)d_in[2 + b];
        const float* Wi  = (const float*)d_in[4 + b];
        const float* mw  = (const float*)d_in[b ? 11 : 6];
        const float* mb  = (const float*)d_in[b ? 12 : 7];
        const float* aw  = (const float*)d_in[b ? 13 : 8];
        const float* ab  = (const float*)d_in[b ? 14 : 9];
        const float* Wat = (const float*)d_in[b ? 15 : 10];

        cvt(ui, ui16, U * I);
        transpose_f32_f16<<<dim3(I / 32, U / 32), 256, 0, stream>>>(ui, uiT16, U, I);
        cvt(Wu, Wu16, D * I);
        cvt(Wi, Wi16, D * U);
        cvt(mw, mw16, I * D);
        cvt(aw, aw16, I * I);
        cvt(Wat, Wat16, U * UI);

        // G1: ue = ui @ Wu.T   [U,D] -> x16 rows [0,U) and catU cols [0,D)
        gemm_wmma<128, 64, 0, 0><<<dim3(U / 128, 1), 256, 0, stream>>>(
            ui16, I, Wu16, I, nullptr,
            x16, D, catU[b], 2 * D, nullptr, 0, U, D, I);

        // G2: ie = ui.T @ Wi.T [I,D] -> x16 rows [U,U+I) and catIE cols [b*D, b*D+D)
        gemm_wmma<128, 64, 0, 0><<<dim3(I / 128, 1), 256, 0, stream>>>(
            uiT16, U, Wi16, U, nullptr,
            x16 + (size_t)U * D, D, catIE + b * D, 2 * D, nullptr, 0, I, D, U);

        // G3: h = tanh(relu(x @ mw.T + mb))   [UI, I]
        gemm_wmma<128, 128, 0, 1><<<dim3(UI / 128, I / 128), 256, 0, stream>>>(
            x16, D, mw16, D, mb,
            h16, I, nullptr, 0, nullptr, 0, UI, I, D);

        // G4: a = tanh(h @ aw.T + ab)         [UI, I]
        gemm_wmma<128, 128, 0, 2><<<dim3(UI / 128, I / 128), 256, 0, stream>>>(
            h16, I, aw16, I, ab,
            a16, I, nullptr, 0, nullptr, 0, UI, I, I);

        // G5: scores = Wat @ a  (B is [K,N])  [U, I] f32
        gemm_wmma<128, 128, 1, 0><<<dim3(U / 128, I / 128), 256, 0, stream>>>(
            Wat16, UI, a16, I, nullptr,
            nullptr, 0, nullptr, 0, scores, I, U, I, UI);

        softmax_rows<<<U, 256, 0, stream>>>(scores, at16, I);

        // G6: ue_at = at @ ie  (B = ie [K,N], rows U.. of x16) -> catU cols [D,2D)
        gemm_wmma<128, 64, 1, 0><<<dim3(U / 128, 1), 256, 0, stream>>>(
            at16, I, x16 + (size_t)U * D, D, nullptr,
            nullptr, 0, catU[b] + D, 2 * D, nullptr, 0, U, D, I);

        // G7: u_b = sigmoid(cat([ue,ue_at]) @ mlp_w.T + mlp_b)  [U, D] f32
        gemm_wmma<128, 64, 0, 3><<<dim3(U / 128, 1), 256, 0, stream>>>(
            catU[b], 2 * D, mlpw16, 2 * D, mlp_b,
            nullptr, 0, nullptr, 0, ufb[b], D, U, D, 2 * D);
    }

    // fusion: rate gating of the two user embeddings
    fuse_users<<<(U * D + 255) / 256, 256, 0, stream>>>(uf0, uf1, fuse16, U * D);

    // item_embeddings = sigmoid(cat([ie_mp,ie_pmp]) @ mlp_w.T + mlp_b) -> out[U*D ..)
    gemm_wmma<128, 64, 0, 3><<<dim3(I / 128, 1), 256, 0, stream>>>(
        catIE, 2 * D, mlpw16, 2 * D, mlp_b,
        nullptr, 0, nullptr, 0, out + (size_t)U * D, D, I, D, 2 * D);

    // user_embeds = tanh(user_fusion @ tr_w.T + tr_b) -> out[0 ..)
    gemm_wmma<128, 64, 0, 2><<<dim3(U / 128, 1), 256, 0, stream>>>(
        fuse16, D, trw16, D, tr_b,
        nullptr, 0, nullptr, 0, out, D, U, D, D);
}
